// MultiHeadSelfAttention_87222195847551
// MI455X (gfx1250) — matrix operand order repaired and checked by reading, not compiled, not run on hardware
//
#include <hip/hip_runtime.h>

typedef __bf16 bf16_t;
typedef __attribute__((ext_vector_type(16))) __bf16 v16bf;
typedef __attribute__((ext_vector_type(8)))  float  v8f;

#define DIM     1024
#define NHEADS  16
#define HDIM    64
#define INNER   1024
#define SEQ     2048
#define NBATCH  4
#define NROWS   (NBATCH * SEQ)     // 8192
#define QKVN    (3 * INNER)        // 3072
#define ATT_SCALE 0.125f           // HDIM^-0.5

// ---------- helpers ----------

__device__ __forceinline__ bf16_t f2bf(float f) {
  unsigned u = __builtin_bit_cast(unsigned, f);
  u += 0x7FFFu + ((u >> 16) & 1u);
  unsigned short h = (unsigned short)(u >> 16);
  return __builtin_bit_cast(bf16_t, h);
}

union FragU { uint4 u[2]; v16bf v; };

__device__ __forceinline__ v16bf load2x16(const bf16_t* p0, const bf16_t* p1) {
  FragU f;
  f.u[0] = *(const uint4*)p0;
  f.u[1] = *(const uint4*)p1;
  return f.v;
}

// A fragment (16x32 bf16), tile row-major [m][k].
// ISA layout: lane m = lane&15; K runs (lane>>4)*8 + {0..7} and +16.
__device__ __forceinline__ v16bf frag_a(const bf16_t* tile, int ld, int mBase, int lane) {
  const bf16_t* p = tile + (size_t)(mBase + (lane & 15)) * ld + ((lane >> 4) << 3);
  return load2x16(p, p + 16);
}

// B fragment (32x16 bf16), tile TRANSPOSED [n][k].
// ISA layout: lane n = lane&15; K runs (lane>>4)*8 + {0..7} and +16.
__device__ __forceinline__ v16bf frag_bT(const bf16_t* tileT, int ld, int nBase, int lane) {
  const bf16_t* p = tileT + (size_t)(nBase + (lane & 15)) * ld + ((lane >> 4) << 3);
  return load2x16(p, p + 16);
}

__device__ __forceinline__ v8f wmma_bf16(v16bf a, v16bf b, v8f c) {
  return __builtin_amdgcn_wmma_f32_16x16x32_bf16(
      false, a, false, b, (short)0, c, false, false);
}

__device__ __forceinline__ v8f zero8() {
  v8f z;
#pragma unroll
  for (int i = 0; i < 8; ++i) z[i] = 0.0f;
  return z;
}

// Async global->LDS copy (ISA 15.18.3 op 98, ASYNCcnt).
__device__ __forceinline__ void async_copy_b128(void* lds, const void* gaddr) {
  unsigned l = (unsigned)(uintptr_t)lds;
  asm volatile("global_load_async_to_lds_b128 %0, %1, off"
               :: "v"(l), "v"(gaddr)
               : "memory");
}
__device__ __forceinline__ void wait_async0() {
  asm volatile("s_wait_asynccnt 0" ::: "memory");
}

// ---------- stage 0: fp32 -> bf16 convert ----------

__global__ void cvt_bf16_kernel(const float* __restrict__ in,
                                bf16_t* __restrict__ out, int n) {
  int i = blockIdx.x * blockDim.x + threadIdx.x;
  if (i < n) out[i] = f2bf(in[i]);
}

// ---------- stages 1 & 3: 128x256 tiled bf16 GEMM, double-buffered ----------
// 8 waves 2(M) x 4(N), 64x64 per wave: 16 WMMAs per 32-wide K step.
// Pipeline: async A + global B loads for tile i+1 issued before computing
// tile i; B registers scattered to LDS after compute; ONE barrier per step.

#define TM 128
#define TN 256
#define TK 32
#define LDAS 40
#define LDBS 40

template <int MODE>
__global__ __launch_bounds__(256) void gemm_bf16_kernel(
    const bf16_t* __restrict__ A, const bf16_t* __restrict__ B,
    int K, int lda, int ldb,
    bf16_t* __restrict__ qO, bf16_t* __restrict__ kO, bf16_t* __restrict__ vO,
    float* __restrict__ out, const float* __restrict__ bias) {
  __shared__ bf16_t As[2][TM * LDAS];   // [m][k]
  __shared__ bf16_t Bs[2][TN * LDBS];   // transposed [n][k]

  const int tid = threadIdx.x;
  const int lane = tid & 31;
  const int wave = tid >> 5;
  const int waveM = wave >> 2;
  const int waveN = wave & 3;
  const int rowBase = blockIdx.y * TM;
  const int colBase = blockIdx.x * TN;

  v8f acc[4][4];
#pragma unroll
  for (int mi = 0; mi < 4; ++mi)
#pragma unroll
    for (int ni = 0; ni < 4; ++ni) acc[mi][ni] = zero8();

  const int arow = tid >> 1;
  const int ahalf = (tid & 1) << 4;
  const int bk = tid >> 3;
  const int bn0 = (tid & 7) << 5;

  const bf16_t* agBase = A + (size_t)(rowBase + arow) * lda + ahalf;
  const bf16_t* bgBase = B + (size_t)bk * ldb + colBase + bn0;
  const int aLds = arow * LDAS + ahalf;

  // ---- prologue: stage K-tile 0 into buffer 0 ----
  {
    async_copy_b128(&As[0][aLds], agBase);
    async_copy_b128(&As[0][aLds + 8], agBase + 8);
    bf16_t btmp[32];
#pragma unroll
    for (int c = 0; c < 4; ++c)
      *(uint4*)&btmp[c * 8] = *(const uint4*)(bgBase + c * 8);
#pragma unroll
    for (int j = 0; j < 32; ++j)
      Bs[0][(bn0 + j) * LDBS + bk] = btmp[j];
  }
  wait_async0();
  __syncthreads();

  int buf = 0;
  for (int kb = 0; kb < K; kb += TK, buf ^= 1) {
    const int nxt = buf ^ 1;
    const bool more = (kb + TK) < K;

    // ---- issue next tile's loads before computing current ----
    bf16_t btmp[32];
    if (more) {
      const bf16_t* ag = agBase + kb + TK;
      async_copy_b128(&As[nxt][aLds], ag);
      async_copy_b128(&As[nxt][aLds + 8], ag + 8);
      const bf16_t* bg = bgBase + (size_t)(kb + TK) * ldb;
#pragma unroll
      for (int c = 0; c < 4; ++c)
        *(uint4*)&btmp[c * 8] = *(const uint4*)(bg + c * 8);
    }

    // ---- 16 WMMAs on current buffer ----
    const bf16_t* Ac = As[buf];
    const bf16_t* Bc = Bs[buf];
    v16bf af[4];
#pragma unroll
    for (int mi = 0; mi < 4; ++mi)
      af[mi] = frag_a(Ac, LDAS, waveM * 64 + mi * 16, lane);
#pragma unroll
    for (int ni = 0; ni < 4; ++ni) {
      v16bf bf = frag_bT(Bc, LDBS, waveN * 64 + ni * 16, lane);
#pragma unroll
      for (int mi = 0; mi < 4; ++mi)
        acc[mi][ni] = wmma_bf16(af[mi], bf, acc[mi][ni]);
    }

    // ---- scatter held B registers into next buffer ----
    if (more) {
#pragma unroll
      for (int j = 0; j < 32; ++j)
        Bs[nxt][(bn0 + j) * LDBS + bk] = btmp[j];
    }
    wait_async0();
    __syncthreads();
  }

  // ---- epilogue (C layout: row m=(lane>>4)*8+r, col n=lane&15) ----
  const int mW = rowBase + waveM * 64 + ((lane >> 4) << 3);
  const int nW = colBase + waveN * 64 + (lane & 15);
#pragma unroll
  for (int mi = 0; mi < 4; ++mi) {
#pragma unroll
    for (int ni = 0; ni < 4; ++ni) {
      const int c = nW + ni * 16;
#pragma unroll
      for (int r = 0; r < 8; ++r) {
        const int row = mW + mi * 16 + r;
        float vval = acc[mi][ni][r];
        if (MODE == 0) {
          const int which = c >> 10;
          const int inner = c & 1023;
          const int h = inner >> 6, d = inner & 63;
          const int batch = row >> 11, seq = row & 2047;
          const size_t dst =
              (((size_t)batch * NHEADS + h) * SEQ + seq) * HDIM + d;
          const bf16_t bv = f2bf(which == 0 ? vval * ATT_SCALE : vval);
          if (which == 0)      qO[dst] = bv;
          else if (which == 1) kO[dst] = bv;
          else                 vO[dst] = bv;
        } else {
          out[(size_t)row * DIM + c] = vval + bias[c];
        }
      }
    }
  }
}

// ---------- stage 2: flash attention, 64-key blocks, double-buffered ----------
// Block: one (b,h) x 128 query rows; wave: 16 rows. Per 64-key block:
// 8 score WMMAs + 8 P@V WMMAs; one cross-lane max reduce covers 64 keys.

#define KBLK 64
#define LDK 72     // Ks [key][d]
#define LDVt 72    // Vt [d][key]
#define LDP 72     // per-wave P patch [16][64]

__global__ __launch_bounds__(256) void attn_kernel(
    const bf16_t* __restrict__ Q, const bf16_t* __restrict__ Km,
    const bf16_t* __restrict__ V, bf16_t* __restrict__ att) {
  __shared__ bf16_t Ks[2][KBLK * LDK];
  __shared__ bf16_t Vt[2][HDIM * LDVt];
  __shared__ bf16_t Ps[8 * 16 * LDP];

  const int bh = blockIdx.x;
  const int qb = blockIdx.y;
  const int tid = threadIdx.x;
  const int lane = tid & 31;
  const int wave = tid >> 5;

  const bf16_t* Qh = Q + (size_t)bh * SEQ * HDIM;
  const bf16_t* Kh = Km + (size_t)bh * SEQ * HDIM;
  const bf16_t* Vh = V + (size_t)bh * SEQ * HDIM;

  const int qRow0 = qb * 128 + wave * 16;

  v16bf qf0, qf1;
  {
    const bf16_t* qp =
        Qh + (size_t)(qRow0 + (lane & 15)) * HDIM + ((lane >> 4) << 3);
    qf0 = load2x16(qp, qp + 16);
    qf1 = load2x16(qp + 32, qp + 48);
  }

  float mrow[8], lpart[8];
  v8f O[4];
#pragma unroll
  for (int r = 0; r < 8; ++r) { mrow[r] = -1e30f; lpart[r] = 0.0f; }
#pragma unroll
  for (int dt = 0; dt < 4; ++dt) O[dt] = zero8();

  // staging map: thread -> (key row, 16-wide d chunk)
  const int lkr = tid >> 2;         // 0..63
  const int ldc = (tid & 3) << 4;   // 0,16,32,48
  const int kLds = lkr * LDK + ldc;
  bf16_t* Pw = Ps + wave * 16 * LDP;

  // ---- prologue: stage key block 0 into buffer 0 ----
  {
    const bf16_t* kg = Kh + (size_t)lkr * HDIM + ldc;
    async_copy_b128(&Ks[0][kLds], kg);
    async_copy_b128(&Ks[0][kLds + 8], kg + 8);
    const bf16_t* vg = Vh + (size_t)lkr * HDIM + ldc;
    bf16_t vtmp[16];
    *(uint4*)&vtmp[0] = *(const uint4*)vg;
    *(uint4*)&vtmp[8] = *(const uint4*)(vg + 8);
#pragma unroll
    for (int j = 0; j < 16; ++j) Vt[0][(ldc + j) * LDVt + lkr] = vtmp[j];
  }
  wait_async0();
  __syncthreads();

  int buf = 0;
  for (int kb0 = 0; kb0 < SEQ; kb0 += KBLK, buf ^= 1) {
    const int nxt = buf ^ 1;
    const bool more = (kb0 + KBLK) < SEQ;

    // ---- issue next block's loads ----
    bf16_t vtmp[16];
    if (more) {
      const bf16_t* kg = Kh + (size_t)(kb0 + KBLK + lkr) * HDIM + ldc;
      async_copy_b128(&Ks[nxt][kLds], kg);
      async_copy_b128(&Ks[nxt][kLds + 8], kg + 8);
      const bf16_t* vg = Vh + (size_t)(kb0 + KBLK + lkr) * HDIM + ldc;
      *(uint4*)&vtmp[0] = *(const uint4*)vg;
      *(uint4*)&vtmp[8] = *(const uint4*)(vg + 8);
    }

    const bf16_t* Kc = Ks[buf];
    const bf16_t* Vc = Vt[buf];

    // ---- scores: 4 key subtiles x 2 d-steps = 8 WMMAs ----
    v8f S[4];
#pragma unroll
    for (int jt = 0; jt < 4; ++jt) {
      const bf16_t* kp0 =
          Kc + (size_t)(jt * 16 + (lane & 15)) * LDK + ((lane >> 4) << 3);
      v16bf bk0 = load2x16(kp0, kp0 + 16);
      v16bf bk1 = load2x16(kp0 + 32, kp0 + 48);
      v8f s = zero8();
      s = wmma_bf16(qf0, bk0, s);
      s = wmma_bf16(qf1, bk1, s);
      S[jt] = s;
    }

    // ---- online softmax: one max reduce per 64 keys; lazy row-sum ----
    float alpha[8];
#pragma unroll
    for (int r = 0; r < 8; ++r) {
      float t = fmaxf(fmaxf(S[0][r], S[1][r]), fmaxf(S[2][r], S[3][r]));
      t = fmaxf(t, __shfl_xor(t, 1));
      t = fmaxf(t, __shfl_xor(t, 2));
      t = fmaxf(t, __shfl_xor(t, 4));
      t = fmaxf(t, __shfl_xor(t, 8));
      const float mn = fmaxf(mrow[r], t);
      alpha[r] = __expf(mrow[r] - mn);
      mrow[r] = mn;
      float psum = 0.0f;
#pragma unroll
      for (int jt = 0; jt < 4; ++jt) {
        const float p = __expf(S[jt][r] - mn);
        S[jt][r] = p;
        psum += p;
      }
      lpart[r] = lpart[r] * alpha[r] + psum;
    }
#pragma unroll
    for (int dt = 0; dt < 4; ++dt)
#pragma unroll
      for (int r = 0; r < 8; ++r) O[dt][r] *= alpha[r];

    // ---- C-layout P -> per-wave LDS patch -> two A fragments ----
    {
      const int mb = (lane >> 4) << 3;
      const int jc = lane & 15;
#pragma unroll
      for (int r = 0; r < 8; ++r) {
#pragma unroll
        for (int jt = 0; jt < 4; ++jt)
          Pw[(mb + r) * LDP + jt * 16 + jc] = f2bf(S[jt][r]);
      }
    }
    asm volatile("" ::: "memory");
    v16bf pf0, pf1;
    {
      const bf16_t* pp = Pw + (size_t)(lane & 15) * LDP + ((lane >> 4) << 3);
      pf0 = load2x16(pp, pp + 16);        // keys 0..31
      pf1 = load2x16(pp + 32, pp + 48);   // keys 32..63
    }

    // ---- O += P @ V : 4 d-tiles x 2 k-steps = 8 WMMAs ----
#pragma unroll
    for (int dt = 0; dt < 4; ++dt) {
      const bf16_t* vp =
          Vc + (size_t)(dt * 16 + (lane & 15)) * LDVt + ((lane >> 4) << 3);
      v16bf vf0 = load2x16(vp, vp + 16);          // keys 0..31
      v16bf vf1 = load2x16(vp + 32, vp + 48);     // keys 32..63
      O[dt] = wmma_bf16(pf0, vf0, O[dt]);
      O[dt] = wmma_bf16(pf1, vf1, O[dt]);
    }

    // ---- scatter held V registers into next buffer ----
    if (more) {
#pragma unroll
      for (int j = 0; j < 16; ++j) Vt[nxt][(ldc + j) * LDVt + lkr] = vtmp[j];
    }
    wait_async0();
    __syncthreads();
  }

  // final row-sum reduce, normalize, store
  float lrow[8];
#pragma unroll
  for (int r = 0; r < 8; ++r) {
    float l = lpart[r];
    l += __shfl_xor(l, 1);
    l += __shfl_xor(l, 2);
    l += __shfl_xor(l, 4);
    l += __shfl_xor(l, 8);
    lrow[r] = l;
  }

  const int b = bh >> 4, h = bh & 15;
  const int mb = (lane >> 4) << 3;
#pragma unroll
  for (int dt = 0; dt < 4; ++dt) {
    const int col = h * HDIM + dt * 16 + (lane & 15);
#pragma unroll
    for (int r = 0; r < 8; ++r) {
      const int seq = qRow0 + mb + r;
      att[((size_t)b * SEQ + seq) * INNER + col] = f2bf(O[dt][r] / lrow[r]);
    }
  }
}

// ---------- launch ----------

extern "C" void kernel_launch(void* const* d_in, const int* in_sizes, int n_in,
                              void* d_out, int out_size, void* d_ws,
                              size_t ws_size, hipStream_t stream) {
  const float* x     = (const float*)d_in[0];
  const float* w_qkv = (const float*)d_in[1];
  const float* w_out = (const float*)d_in[2];
  const float* b_out = (const float*)d_in[3];
  float* out = (float*)d_out;

  char* ws = (char*)d_ws;
  bf16_t* xb    = (bf16_t*)ws; ws += (size_t)NROWS * DIM * 2;
  bf16_t* wqkvb = (bf16_t*)ws; ws += (size_t)DIM * QKVN * 2;
  bf16_t* woutb = (bf16_t*)ws; ws += (size_t)INNER * DIM * 2;
  bf16_t* qb    = (bf16_t*)ws; ws += (size_t)NROWS * INNER * 2;
  bf16_t* kb    = (bf16_t*)ws; ws += (size_t)NROWS * INNER * 2;
  bf16_t* vb    = (bf16_t*)ws; ws += (size_t)NROWS * INNER * 2;
  bf16_t* attb  = (bf16_t*)ws; ws += (size_t)NROWS * INNER * 2;

  {
    int n0 = NROWS * DIM;
    cvt_bf16_kernel<<<(n0 + 255) / 256, 256, 0, stream>>>(x, xb, n0);
    int n1 = DIM * QKVN;
    cvt_bf16_kernel<<<(n1 + 255) / 256, 256, 0, stream>>>(w_qkv, wqkvb, n1);
    int n2 = INNER * DIM;
    cvt_bf16_kernel<<<(n2 + 255) / 256, 256, 0, stream>>>(w_out, woutb, n2);
  }

  gemm_bf16_kernel<0><<<dim3(QKVN / TN, NROWS / TM), 256, 0, stream>>>(
      xb, wqkvb, DIM, DIM, QKVN, qb, kb, vb, nullptr, nullptr);

  attn_kernel<<<dim3(NBATCH * NHEADS, SEQ / 128), 256, 0, stream>>>(
      qb, kb, vb, attb);

  gemm_bf16_kernel<1><<<dim3(DIM / TN, NROWS / TM), 256, 0, stream>>>(
      attb, woutb, INNER, INNER, DIM, nullptr, nullptr, nullptr, out, b_out);
}
